// Linear_18330920419817
// MI455X (gfx1250) — 8-bit matrix operand order repaired and checked by reading, compiled, not run on hardware
//
#include <hip/hip_runtime.h>
#include <stdint.h>


typedef __attribute__((ext_vector_type(8))) int v8i;
typedef __attribute__((ext_vector_type(4))) int vi4;

#define M_DIM 8192
#define K_DIM 4096
#define N_DIM 4096

#define TILE_M 128
#define TILE_N 128
#define TILE_K 128
#define LDS_STRIDE 144

typedef __attribute__((address_space(1))) vi4 gvi4;
typedef __attribute__((address_space(3))) vi4 lvi4;

__device__ __forceinline__ lvi4* lds_v4(void* p) {
    return (lvi4*)(uint32_t)(uintptr_t)p;
}
__device__ __forceinline__ gvi4* glb_v4(const void* p) {
    return (gvi4*)(uintptr_t)p;
}

union Frag {
    v8i   v;
    uint2 u2[4];
    uint4 u4[2];
};


__device__ __forceinline__ void stage_a_async(const int8_t* __restrict__ A,
                                              uint8_t* sA, int m0, int k0,
                                              int arow, int ahalf) {
    const uint8_t* g = (const uint8_t*)A + (size_t)(m0 + arow) * K_DIM + k0 + ahalf;
    uint8_t* l = sA + arow * LDS_STRIDE + ahalf;
#pragma unroll
    for (int t = 0; t < 4; ++t)
        __builtin_amdgcn_global_load_async_to_lds_b128(
            glb_v4(g + 16 * t), lds_v4(l + 16 * t), 0, 0);
}

__device__ __forceinline__ void load_b_regs(const int8_t* __restrict__ W,
                                            int k0, int n0, int bkb, int bn4,
                                            uint32_t r[16]) {
#pragma unroll
    for (int i = 0; i < 4; ++i) {
        const uint8_t* g = (const uint8_t*)W + (size_t)(k0 + bkb + 4 * i) * N_DIM + n0 + bn4;
#pragma unroll
        for (int j = 0; j < 4; ++j)
            r[4 * i + j] = *(const uint32_t*)(g + (size_t)j * N_DIM);
    }
}

__device__ __forceinline__ void store_bt_lds(uint8_t* sBT, int bkb, int bn4,
                                             const uint32_t r[16]) {
#pragma unroll
    for (int i = 0; i < 4; ++i) {
        uint32_t r0 = r[4 * i + 0], r1 = r[4 * i + 1];
        uint32_t r2 = r[4 * i + 2], r3 = r[4 * i + 3];
        uint32_t a01l = __builtin_amdgcn_perm(r1, r0, 0x05010400u);
        uint32_t a01h = __builtin_amdgcn_perm(r1, r0, 0x07030602u);
        uint32_t a23l = __builtin_amdgcn_perm(r3, r2, 0x05010400u);
        uint32_t a23h = __builtin_amdgcn_perm(r3, r2, 0x07030602u);
        uint8_t* dB = sBT + bkb + 4 * i;
        *(uint32_t*)(dB + (size_t)(bn4 + 0) * LDS_STRIDE) =
            __builtin_amdgcn_perm(a23l, a01l, 0x05040100u);
        *(uint32_t*)(dB + (size_t)(bn4 + 1) * LDS_STRIDE) =
            __builtin_amdgcn_perm(a23l, a01l, 0x07060302u);
        *(uint32_t*)(dB + (size_t)(bn4 + 2) * LDS_STRIDE) =
            __builtin_amdgcn_perm(a23h, a01h, 0x05040100u);
        *(uint32_t*)(dB + (size_t)(bn4 + 3) * LDS_STRIDE) =
            __builtin_amdgcn_perm(a23h, a01h, 0x07060302u);
    }
}


__global__ __launch_bounds__(256)
void int8_gemm_dequant(const int8_t* __restrict__ A,
                       const float*  __restrict__ a_s,
                       const int8_t* __restrict__ W,
                       const float*  __restrict__ w_s,
                       float*        __restrict__ out)
{
    __shared__ __attribute__((aligned(16))) uint8_t sA [2][TILE_M * LDS_STRIDE];
    __shared__ __attribute__((aligned(16))) uint8_t sBT[2][TILE_N * LDS_STRIDE];

    const int tid   = threadIdx.x;
    const int lane  = tid & 31;
    const int wave  = tid >> 5;
    const int wm    = wave >> 1;
    const int wn    = wave & 1;
    const int lrow  = lane & 15;
    const int lhalf = lane >> 4;

    const int m0 = blockIdx.y * TILE_M;
    const int n0 = blockIdx.x * TILE_N;

    const int arow  = tid >> 1;
    const int ahalf = (tid & 1) * 64;
    const int bn4   = (tid & 31) * 4;
    const int bkb   = (tid >> 5) * 16;

    v8i acc[2][4];
#pragma unroll
    for (int i = 0; i < 2; ++i)
#pragma unroll
        for (int j = 0; j < 4; ++j)
            acc[i][j] = (v8i){0, 0, 0, 0, 0, 0, 0, 0};

    {
        stage_a_async(A, sA[0], m0, 0, arow, ahalf);
        uint32_t r[16];
        load_b_regs(W, 0, n0, bkb, bn4, r);
        store_bt_lds(sBT[0], bkb, bn4, r);
        asm volatile("s_wait_asynccnt 0" ::: "memory");
        __syncthreads();
    }

    int buf = 0;
    for (int k0 = 0; k0 < K_DIM; k0 += TILE_K) {
        const bool has_next = (k0 + TILE_K) < K_DIM;
        uint32_t rnext[16];

        if (has_next) {
            stage_a_async(A, sA[buf ^ 1], m0, k0 + TILE_K, arow, ahalf);
            load_b_regs(W, k0 + TILE_K, n0, bkb, bn4, rnext);
        }

        const uint8_t* cA = sA[buf];
        const uint8_t* cB = sBT[buf];
#pragma unroll
        for (int ks = 0; ks < TILE_K; ks += 64) {
            Frag af0, af1, bf0, bf1, bf2, bf3;
            {
                const uint8_t* b0 = cA + (size_t)(wm * 32 + lrow) * LDS_STRIDE + ks + lhalf * 16;
                const uint8_t* b1 = b0 + 16 * LDS_STRIDE;
                af0.u2[0] = *(const uint2*)(b0 +  0);
                af0.u2[1] = *(const uint2*)(b0 +  8);
                af0.u2[2] = *(const uint2*)(b0 + 32);
                af0.u2[3] = *(const uint2*)(b0 + 40);
                af1.u2[0] = *(const uint2*)(b1 +  0);
                af1.u2[1] = *(const uint2*)(b1 +  8);
                af1.u2[2] = *(const uint2*)(b1 + 32);
                af1.u2[3] = *(const uint2*)(b1 + 40);
            }
            {
                const uint8_t* bb = cB + (size_t)(wn * 64 + lrow) * LDS_STRIDE + ks + lhalf * 16;
                bf0.u4[0] = *(const uint4*)(bb);
                bf0.u4[1] = *(const uint4*)(bb + 32);
                bb += 16 * LDS_STRIDE;
                bf1.u4[0] = *(const uint4*)(bb);
                bf1.u4[1] = *(const uint4*)(bb + 32);
                bb += 16 * LDS_STRIDE;
                bf2.u4[0] = *(const uint4*)(bb);
                bf2.u4[1] = *(const uint4*)(bb + 32);
                bb += 16 * LDS_STRIDE;
                bf3.u4[0] = *(const uint4*)(bb);
                bf3.u4[1] = *(const uint4*)(bb + 32);
            }
            acc[0][0] = __builtin_amdgcn_wmma_i32_16x16x64_iu8(true, af0.v, true, bf0.v, acc[0][0], false, false);
            acc[1][0] = __builtin_amdgcn_wmma_i32_16x16x64_iu8(true, af1.v, true, bf0.v, acc[1][0], false, false);
            acc[0][1] = __builtin_amdgcn_wmma_i32_16x16x64_iu8(true, af0.v, true, bf1.v, acc[0][1], false, false);
            acc[1][1] = __builtin_amdgcn_wmma_i32_16x16x64_iu8(true, af1.v, true, bf1.v, acc[1][1], false, false);
            acc[0][2] = __builtin_amdgcn_wmma_i32_16x16x64_iu8(true, af0.v, true, bf2.v, acc[0][2], false, false);
            acc[1][2] = __builtin_amdgcn_wmma_i32_16x16x64_iu8(true, af1.v, true, bf2.v, acc[1][2], false, false);
            acc[0][3] = __builtin_amdgcn_wmma_i32_16x16x64_iu8(true, af0.v, true, bf3.v, acc[0][3], false, false);
            acc[1][3] = __builtin_amdgcn_wmma_i32_16x16x64_iu8(true, af1.v, true, bf3.v, acc[1][3], false, false);
        }

        if (has_next)
            store_bt_lds(sBT[buf ^ 1], bkb, bn4, rnext);

        asm volatile("s_wait_asynccnt 0" ::: "memory");
        __syncthreads();
        buf ^= 1;
    }

    float wsv[4];
#pragma unroll
    for (int jn = 0; jn < 4; ++jn)
        wsv[jn] = w_s[n0 + wn * 64 + jn * 16 + lrow];

#pragma unroll
    for (int i = 0; i < 2; ++i) {
        const int mbase = m0 + wm * 32 + i * 16 + lhalf * 8;
#pragma unroll
        for (int e = 0; e < 8; ++e) {
            const int   m  = mbase + e;
            const float as = a_s[m];
#pragma unroll
            for (int jn = 0; jn < 4; ++jn) {
                const int n = n0 + wn * 64 + jn * 16 + lrow;
                out[(size_t)m * N_DIM + n] = (float)acc[i][jn][e] * as * wsv[jn];
            }
        }
    }
}

extern "C" void kernel_launch(void* const* d_in, const int* in_sizes, int n_in,
                              void* d_out, int out_size, void* d_ws, size_t ws_size,
                              hipStream_t stream) {
    (void)in_sizes; (void)n_in; (void)out_size; (void)d_ws; (void)ws_size;
    const int8_t* a   = (const int8_t*)d_in[0];
    const float*  a_s = (const float*) d_in[1];
    const int8_t* w   = (const int8_t*)d_in[2];
    const float*  w_s = (const float*) d_in[3];
    float* out = (float*)d_out;

    dim3 grid(N_DIM / TILE_N, M_DIM / TILE_M);
    int8_gemm_dequant<<<grid, 256, 0, stream>>>(a, a_s, w, w_s, out);
}
